// Step3p5Experts_6674379178226
// MI455X (gfx1250) — matrix operand order repaired and checked by reading, not compiled, not run on hardware
//
#include <hip/hip_runtime.h>
#include <hip/hip_bf16.h>

// ---------------------------------------------------------------------------
// MoE (Step3.5-style experts) for gfx1250 / MI455X.
// bf16 WMMA path: v_wmma_f32_16x16x32_bf16, fp32 accumulate.
// 32x32 register tiles per wave for 2-4x L2 traffic reduction vs 16x16.
// ---------------------------------------------------------------------------

typedef __bf16 bf16_t;
typedef __attribute__((ext_vector_type(16))) __bf16 v16bf;
typedef __attribute__((ext_vector_type(8)))  __bf16 v8bf;
typedef __attribute__((ext_vector_type(8)))  float  v8f;

constexpr int E    = 8;
constexpr int H    = 1024;
constexpr int I    = 512;
constexpr int TOPK = 2;
constexpr int T    = 2048;          // B*S
constexpr int KE   = E * I;         // 4096 fused K for GEMM2

// --------------------------- fragment loaders ------------------------------
// A (16x32 bf16, MxK), row-major memory, caller passes &A[row(lane)][kb].
// lane<16 : halves = K[0..7]  then K[16..23]
// lane>=16: halves = K[8..15] then K[24..31]
__device__ __forceinline__ v16bf load_a_frag(const bf16_t* __restrict__ p, int lane) {
  const int off = (lane & 16) ? 8 : 0;
  union { v16bf v; v8bf h[2]; } u;
  u.h[0] = *(const v8bf*)(p + off);
  u.h[1] = *(const v8bf*)(p + off + 16);
  return u.v;
}

// B (32x16 bf16, KxN), column-major memory (K contiguous per column),
// caller passes &B[col(lane)][kb].
// lane<16 : K[0..7] then K[16..23] ; lane>=16 : K[8..15] then K[24..31]
__device__ __forceinline__ v16bf load_b_frag(const bf16_t* __restrict__ p, int lane) {
  const int off = (lane & 16) ? 8 : 0;
  union { v16bf v; v8bf h[2]; } u;
  u.h[0] = *(const v8bf*)(p + off);
  u.h[1] = *(const v8bf*)(p + off + 16);
  return u.v;
}

// --------------------------- prep kernels ----------------------------------
__global__ void cvt_f32_bf16(const float* __restrict__ in, bf16_t* __restrict__ out, int n) {
  int i = blockIdx.x * blockDim.x + threadIdx.x;
  if (i < n) out[i] = (bf16_t)in[i];
}

// in: [batch][R][C] fp32, out: batched bf16 with out[c*out_ld + r]
__global__ void transpose_cvt(const float* __restrict__ in, bf16_t* __restrict__ out,
                              int R, int C, long long in_bstride,
                              long long out_bstride, int out_ld) {
  __shared__ float tile[32][33];
  const int b  = blockIdx.z;
  const int c0 = blockIdx.x * 32;
  const int r0 = blockIdx.y * 32;
  const float* inp = in + (long long)b * in_bstride;
  bf16_t* outp     = out + (long long)b * out_bstride;
  const int tx = threadIdx.x, ty = threadIdx.y;   // 32 x 8
  #pragma unroll
  for (int i = ty; i < 32; i += 8)
    tile[i][tx] = inp[(long long)(r0 + i) * C + (c0 + tx)];
  __syncthreads();
  #pragma unroll
  for (int i = ty; i < 32; i += 8)
    outp[(long long)(c0 + i) * out_ld + (r0 + tx)] = (bf16_t)tile[tx][i];
}

__global__ void build_combine(const int* __restrict__ idx, const float* __restrict__ w,
                              float* __restrict__ comb) {
  int t = blockIdx.x * blockDim.x + threadIdx.x;   // t < T
  float c[E];
  #pragma unroll
  for (int e = 0; e < E; ++e) c[e] = 0.0f;
  #pragma unroll
  for (int k = 0; k < TOPK; ++k) {
    int e = idx[t * TOPK + k];
    if (e >= 0 && e < E) c[e] += w[t * TOPK + k];
  }
  #pragma unroll
  for (int e = 0; e < E; ++e) comb[t * E + e] = c[e];
}

// --------------------------- GEMM1 + SwiGLU --------------------------------
// Per wave: 32(tokens) x 32(i-cols) tile => 8 accumulators (4 gate + 4 up),
// 6 fragment loads per K-step feeding 8 WMMAs.
// xb:   [T][H] bf16 (row-major)
// gut:  [E][2I][H] bf16 (column n contiguous over k=H)
// comb: [T][E] fp32
// inter:[T][KE] bf16, inter[t][e*I+i] = comb[t][e]*silu(gate)*up
__global__ void moe_gemm1(const bf16_t* __restrict__ xb, const bf16_t* __restrict__ gut,
                          const float* __restrict__ comb, bf16_t* __restrict__ inter) {
  const int lane = threadIdx.x & 31;
  const int gw   = (blockIdx.x * blockDim.x + threadIdx.x) >> 5;
  constexpr int tilesT = T / 32;        // 64
  constexpr int tilesI = I / 32;        // 16
  constexpr int perE   = tilesT * tilesI;
  const int e  = gw / perE;
  const int r  = gw % perE;
  const int tm = r / tilesI;
  const int tn = r % tilesI;

  const int row0 = tm * 32 + (lane & 15);
  const bf16_t* ap0 = xb + (size_t)row0 * H;
  const bf16_t* ap1 = ap0 + (size_t)16 * H;

  const int col0 = tn * 32 + (lane & 15);
  const bf16_t* bg0 = gut + ((size_t)e * (2 * I) + col0) * H;   // gate cols
  const bf16_t* bg1 = bg0 + (size_t)16 * H;
  const bf16_t* bu0 = bg0 + (size_t)I * H;                      // up cols
  const bf16_t* bu1 = bu0 + (size_t)16 * H;

  v8f aG[2][2] = {};   // [am][bn] gate accumulators
  v8f aU[2][2] = {};   // [am][bn] up   accumulators

  for (int kb = 0; kb < H; kb += 32) {
    v16bf a0 = load_a_frag(ap0 + kb, lane);
    v16bf a1 = load_a_frag(ap1 + kb, lane);
    v16bf g0 = load_b_frag(bg0 + kb, lane);
    v16bf g1 = load_b_frag(bg1 + kb, lane);
    v16bf u0 = load_b_frag(bu0 + kb, lane);
    v16bf u1 = load_b_frag(bu1 + kb, lane);
    aG[0][0] = __builtin_amdgcn_wmma_f32_16x16x32_bf16(false, a0, false, g0, (short)0, aG[0][0], false, false);
    aG[0][1] = __builtin_amdgcn_wmma_f32_16x16x32_bf16(false, a0, false, g1, (short)0, aG[0][1], false, false);
    aG[1][0] = __builtin_amdgcn_wmma_f32_16x16x32_bf16(false, a1, false, g0, (short)0, aG[1][0], false, false);
    aG[1][1] = __builtin_amdgcn_wmma_f32_16x16x32_bf16(false, a1, false, g1, (short)0, aG[1][1], false, false);
    aU[0][0] = __builtin_amdgcn_wmma_f32_16x16x32_bf16(false, a0, false, u0, (short)0, aU[0][0], false, false);
    aU[0][1] = __builtin_amdgcn_wmma_f32_16x16x32_bf16(false, a0, false, u1, (short)0, aU[0][1], false, false);
    aU[1][0] = __builtin_amdgcn_wmma_f32_16x16x32_bf16(false, a1, false, u0, (short)0, aU[1][0], false, false);
    aU[1][1] = __builtin_amdgcn_wmma_f32_16x16x32_bf16(false, a1, false, u1, (short)0, aU[1][1], false, false);
  }

  #pragma unroll
  for (int am = 0; am < 2; ++am) {
    const int mbase = tm * 32 + am * 16 + ((lane & 16) ? 8 : 0);
    #pragma unroll
    for (int bn = 0; bn < 2; ++bn) {
      const int n = tn * 32 + bn * 16 + (lane & 15);
      #pragma unroll
      for (int v = 0; v < 8; ++v) {
        const int m = mbase + v;
        const float w  = comb[m * E + e];
        const float gv = aG[am][bn][v];
        const float uv = aU[am][bn][v];
        const float s  = gv / (1.0f + __expf(-gv));   // silu
        inter[(size_t)m * KE + e * I + n] = (bf16_t)(w * s * uv);
      }
    }
  }
}

// --------------------------- GEMM2 (fused down) ----------------------------
// Per wave: 32x32 tile => 4 accumulators, 4 fragment loads per K-step.
// inter:[T][KE] bf16 row-major ; dnt:[H][KE] bf16 (column h contiguous over k)
// out:  [T][H] fp32
__global__ void moe_gemm2(const bf16_t* __restrict__ inter, const bf16_t* __restrict__ dnt,
                          float* __restrict__ out) {
  const int lane = threadIdx.x & 31;
  const int gw   = (blockIdx.x * blockDim.x + threadIdx.x) >> 5;
  constexpr int tilesH = H / 32;        // 32
  const int tm = gw / tilesH;
  const int tn = gw % tilesH;

  const int row0 = tm * 32 + (lane & 15);
  const bf16_t* ap0 = inter + (size_t)row0 * KE;
  const bf16_t* ap1 = ap0 + (size_t)16 * KE;

  const int col0 = tn * 32 + (lane & 15);
  const bf16_t* bp0 = dnt + (size_t)col0 * KE;
  const bf16_t* bp1 = bp0 + (size_t)16 * KE;

  v8f acc[2][2] = {};
  for (int kb = 0; kb < KE; kb += 32) {
    v16bf a0 = load_a_frag(ap0 + kb, lane);
    v16bf a1 = load_a_frag(ap1 + kb, lane);
    v16bf b0 = load_b_frag(bp0 + kb, lane);
    v16bf b1 = load_b_frag(bp1 + kb, lane);
    acc[0][0] = __builtin_amdgcn_wmma_f32_16x16x32_bf16(false, a0, false, b0, (short)0, acc[0][0], false, false);
    acc[0][1] = __builtin_amdgcn_wmma_f32_16x16x32_bf16(false, a0, false, b1, (short)0, acc[0][1], false, false);
    acc[1][0] = __builtin_amdgcn_wmma_f32_16x16x32_bf16(false, a1, false, b0, (short)0, acc[1][0], false, false);
    acc[1][1] = __builtin_amdgcn_wmma_f32_16x16x32_bf16(false, a1, false, b1, (short)0, acc[1][1], false, false);
  }

  #pragma unroll
  for (int am = 0; am < 2; ++am) {
    const int mbase = tm * 32 + am * 16 + ((lane & 16) ? 8 : 0);
    #pragma unroll
    for (int bn = 0; bn < 2; ++bn) {
      const int n = tn * 32 + bn * 16 + (lane & 15);
      #pragma unroll
      for (int v = 0; v < 8; ++v)
        out[(size_t)(mbase + v) * H + n] = acc[am][bn][v];
    }
  }
}

// --------------------------- launcher --------------------------------------
extern "C" void kernel_launch(void* const* d_in, const int* in_sizes, int n_in,
                              void* d_out, int out_size, void* d_ws, size_t ws_size,
                              hipStream_t stream) {
  const float* hs   = (const float*)d_in[0];   // [T][H]
  const int*   ridx = (const int*)d_in[1];     // [T][K]
  const float* rw   = (const float*)d_in[2];   // [T][K]
  const float* gup  = (const float*)d_in[3];   // [E][H][2I]
  const float* dwn  = (const float*)d_in[4];   // [E][I][H]
  float* out = (float*)d_out;

  char* ws = (char*)d_ws;
  size_t off = 0;
  auto take = [&](size_t bytes) -> char* {
    char* p = ws + off;
    off += (bytes + 255) & ~(size_t)255;
    return p;
  };
  bf16_t* xb    = (bf16_t*)take((size_t)T * H * 2);            //  4 MB
  bf16_t* gut   = (bf16_t*)take((size_t)E * 2 * I * H * 2);    // 16 MB
  bf16_t* dnt   = (bf16_t*)take((size_t)H * KE * 2);           //  8 MB
  bf16_t* inter = (bf16_t*)take((size_t)T * KE * 2);           // 16 MB
  float*  comb  = (float*)take((size_t)T * E * 4);             // 64 KB

  // 1) x -> bf16
  cvt_f32_bf16<<<(T * H) / 256, 256, 0, stream>>>(hs, xb, T * H);

  // 2) gate_up_proj [e][h][f] -> gut [e][f][h]  (R=H, C=2I, out_ld=H)
  {
    dim3 tb(32, 8);
    dim3 gr((2 * I) / 32, H / 32, E);
    transpose_cvt<<<gr, tb, 0, stream>>>(gup, gut, H, 2 * I,
                                         (long long)H * 2 * I,
                                         (long long)2 * I * H, H);
  }
  // 3) down_proj [e][i][h] -> dnt [h][e*I+i]  (R=I, C=H, out_ld=KE, batch off=I)
  {
    dim3 tb(32, 8);
    dim3 gr(H / 32, I / 32, E);
    transpose_cvt<<<gr, tb, 0, stream>>>(dwn, dnt, I, H,
                                         (long long)I * H,
                                         (long long)I, KE);
  }
  // 4) combine weights
  build_combine<<<T / 256, 256, 0, stream>>>(ridx, rw, comb);

  // 5) GEMM1 + SwiGLU + combine: waves = E*(T/32)*(I/32) = 8192 -> 1024 blocks
  moe_gemm1<<<(E * (T / 32) * (I / 32)) / 8, 256, 0, stream>>>(xb, gut, comb, inter);

  // 6) GEMM2 fused over experts: waves = (T/32)*(H/32) = 2048 -> 256 blocks
  moe_gemm2<<<((T / 32) * (H / 32)) / 8, 256, 0, stream>>>(inter, dnt, out);
}
